// GraphGatedEdgeNet_6373731467576
// MI455X (gfx1250) — matrix operand order repaired and checked by reading, compiled, not run on hardware
//
#include <hip/hip_runtime.h>

#define NN 20000
#define NE 320000
#define CC 128

typedef __bf16 bf16_t;
typedef bf16_t v16bf __attribute__((ext_vector_type(16)));
typedef float  v8f   __attribute__((ext_vector_type(8)));

union BF16Frag {
    v16bf v;
    unsigned short u[16];
    uint4 q[2];
};

// Hardware bf16 convert (lowers to v_cvt_pk_bf16_f32 for pairs)
__device__ __forceinline__ unsigned short f2bf(float f) {
    bf16_t b = (bf16_t)f;
    return __builtin_bit_cast(unsigned short, b);
}

__device__ __forceinline__ v8f wmma_bf16(v16bf a, v16bf b, v8f c) {
    return __builtin_amdgcn_wmma_f32_16x16x32_bf16(false, a, false, b, (short)0, c, false, false);
}

// ---------------------------------------------------------------------------
// Prep kernels
// ---------------------------------------------------------------------------

__global__ void deg_kernel(const int* __restrict__ dst, float* __restrict__ deg, int E) {
    int e = blockIdx.x * blockDim.x + threadIdx.x;
    if (e < E) atomicAdd(&deg[dst[e]], 1.0f);
}

__global__ void invdeg_kernel(const float* __restrict__ deg, float* __restrict__ invd, int n) {
    int i = blockIdx.x * blockDim.x + threadIdx.x;
    if (i < n) invd[i] = 1.0f / fmaxf(deg[i], 1.0f);
}

// Pack a f32 [K x N] matrix into WMMA B-fragment order (bf16).
// transposed==0: B[k][n] = src[k*N + n]; transposed==1: B[k][n] = src[n*K + k].
// Fragment order: dst[((kt*nTiles + nt)*32 + lane)*16 + j],
//   k = kt*32 + (lane>>4)*16 + j,  n = nt*16 + (lane&15)
__global__ void pack_b_kernel(const float* __restrict__ src, unsigned short* __restrict__ dst,
                              int K, int N, int transposed) {
    int idx = blockIdx.x * blockDim.x + threadIdx.x;
    if (idx >= K * N) return;
    int j    = idx & 15;
    int lane = (idx >> 4) & 31;
    int frag = idx >> 9;
    int nTiles = N >> 4;
    int nt = frag % nTiles;
    int kt = frag / nTiles;
    int k  = (kt << 5) + ((lane >> 4) << 4) + j;
    int n  = (nt << 4) + (lane & 15);
    float v = transposed ? src[n * K + k] : src[k * N + n];
    dst[idx] = f2bf(v);
}

// Fuse BatchNorm (+ optional preceding bias) into per-channel scale/shift:
//   BN(y + b) = y*s + t,  s = gamma*rsqrt(var+eps),  t = (b - mean)*s + beta
__global__ void bnfuse_kernel(const float* __restrict__ gamma, const float* __restrict__ beta,
                              const float* __restrict__ mean,  const float* __restrict__ var,
                              const float* __restrict__ bias,
                              float* __restrict__ s, float* __restrict__ t, int n) {
    int i = blockIdx.x * blockDim.x + threadIdx.x;
    if (i >= n) return;
    float sc = gamma[i] * rsqrtf(var[i] + 1e-5f);
    float b  = bias ? bias[i] : 0.0f;
    s[i] = sc;
    t[i] = (b - mean[i]) * sc + beta[i];
}

// ---------------------------------------------------------------------------
// Generic WMMA GEMM:  C[M x N] = (A[M x K] * rowScale?) @ B  * s[n] + t[n]
// Block = 256 threads (8 waves), covers a 16-row tile of A (staged bf16 in LDS).
// Wave w computes column tiles nt = w, w+8, ... (TPW tiles; requires N == TPW*128)
// ---------------------------------------------------------------------------
template <int TPW>
__global__ void __launch_bounds__(256) wmma_gemm_kernel(
        const float* __restrict__ A, const float* __restrict__ rowScale,
        const unsigned short* __restrict__ Bp,
        const float* __restrict__ s, const float* __restrict__ t,
        float* __restrict__ C, int M, int K, int N) {
    extern __shared__ unsigned short aLds[];   // 16 * K bf16
    const int tid  = threadIdx.x;
    const int lane = tid & 31;
    const int wave = tid >> 5;
    const int rowBase = blockIdx.x << 4;

    for (int idx = tid; idx < (K << 4); idx += 256) {
        int r = idx / K, k = idx - r * K;
        int row = rowBase + r;
        float v = 0.0f;
        if (row < M) {
            v = A[(size_t)row * K + k];
            if (rowScale) v *= rowScale[row];
        }
        aLds[r * K + k] = f2bf(v);
    }
    __syncthreads();

    const int nTiles = N >> 4;
    const int kTiles = K >> 5;
    const int m  = lane & 15;
    const int kb = (lane >> 4) << 4;

    v8f acc[TPW];
#pragma unroll
    for (int i = 0; i < TPW; ++i)
#pragma unroll
        for (int q = 0; q < 8; ++q) acc[i][q] = 0.0f;

    for (int kt = 0; kt < kTiles; ++kt) {
        const int k0 = (kt << 5) + kb;
        BF16Frag af;
        af.q[0] = *(const uint4*)(aLds + m * K + k0);        // 8 bf16 (16B)
        af.q[1] = *(const uint4*)(aLds + m * K + k0 + 8);   // 8 bf16 (16B)
        // preload all B fragments for this K-step, then run WMMAs back-to-back
        BF16Frag bfr[TPW];
#pragma unroll
        for (int i = 0; i < TPW; ++i) {
            const int nt = wave + (i << 3);
            const uint4* bp = (const uint4*)(Bp + (((size_t)(kt * nTiles + nt)) << 9) + (lane << 4));
            bfr[i].q[0] = bp[0];
            bfr[i].q[1] = bp[1];
        }
#pragma unroll
        for (int i = 0; i < TPW; ++i)
            acc[i] = wmma_bf16(af.v, bfr[i].v, acc[i]);
    }

    const int col = lane & 15;
    const int r0  = (lane >> 4) << 3;
#pragma unroll
    for (int i = 0; i < TPW; ++i) {
        const int n  = ((wave + (i << 3)) << 4) + col;
        const float sc = s ? s[n] : 1.0f;
        const float sh = t ? t[n] : 0.0f;
#pragma unroll
        for (int r = 0; r < 8; ++r) {
            const int row = rowBase + r0 + r;
            if (row < M) C[(size_t)row * N + n] = acc[i][r] * sc + sh;
        }
    }
}

// ---------------------------------------------------------------------------
// Fused EdgeConv: per edge msg = ReLU(BN2(W2 @ ReLU(BN1(W1 @ [xi | xj-xi]))))
// scattered (sum) into agg[dst]. 128 edges/block, 16 edges/wave.
// LDS: W1 packed bf16 (64KB) + W2 packed bf16 (32KB) + per-wave h1 (32KB).
// ---------------------------------------------------------------------------
__global__ void __launch_bounds__(256) edgeconv_kernel(
        const float* __restrict__ xin, const int* __restrict__ src, const int* __restrict__ dst,
        const unsigned short* __restrict__ w1p, const float* __restrict__ s1, const float* __restrict__ t1,
        const unsigned short* __restrict__ w2p, const float* __restrict__ s2, const float* __restrict__ t2,
        float* __restrict__ agg, int E) {
    extern __shared__ unsigned short lds[];
    unsigned short* w1l = lds;              // 32768 ushorts (256x128 packed)
    unsigned short* w2l = lds + 32768;      // 16384 ushorts (128x128 packed)
    unsigned short* h1l = lds + 49152;      // 8 waves * 16*128

    const int tid  = threadIdx.x;
    const int lane = tid & 31;
    const int wave = tid >> 5;

    {   // stage packed weights into LDS (vectorized, 16B per op)
        const uint4* w1s = (const uint4*)w1p; uint4* w1d = (uint4*)w1l;
        for (int i = tid; i < 4096; i += 256) w1d[i] = w1s[i];
        const uint4* w2s = (const uint4*)w2p; uint4* w2d = (uint4*)w2l;
        for (int i = tid; i < 2048; i += 256) w2d[i] = w2s[i];
    }
    __syncthreads();

    const int eBase = blockIdx.x * 128 + wave * 16;
    const int m   = lane & 15;
    const int kb  = (lane >> 4) << 4;
    const int col = lane & 15;
    const int r0  = (lane >> 4) << 3;

    const int e  = eBase + m;
    const int ec = (e < E) ? e : (E - 1);
    const float* xs = xin + (size_t)src[ec] * CC;
    const float* xd = xin + (size_t)dst[ec] * CC;

    // ---- fc1: [16 x 256] @ [256 x 128] ----
    v8f acc[8];
#pragma unroll
    for (int i = 0; i < 8; ++i)
#pragma unroll
        for (int q = 0; q < 8; ++q) acc[i][q] = 0.0f;

#pragma unroll
    for (int kt = 0; kt < 8; ++kt) {
        const int k0 = (kt << 5) + kb;   // 32B-aligned float4 groups
        BF16Frag af;
        if (kt < 4) {                    // K < 128: A = xi  (uniform branch)
            const float4 a0 = *(const float4*)(xd + k0);
            const float4 a1 = *(const float4*)(xd + k0 + 4);
            const float4 b0 = *(const float4*)(xd + k0 + 8);
            const float4 b1 = *(const float4*)(xd + k0 + 12);
            af.u[0] = f2bf(a0.x); af.u[1] = f2bf(a0.y); af.u[2] = f2bf(a0.z); af.u[3] = f2bf(a0.w);
            af.u[4] = f2bf(a1.x); af.u[5] = f2bf(a1.y); af.u[6] = f2bf(a1.z); af.u[7] = f2bf(a1.w);
            af.u[8]  = f2bf(b0.x); af.u[9]  = f2bf(b0.y); af.u[10] = f2bf(b0.z); af.u[11] = f2bf(b0.w);
            af.u[12] = f2bf(b1.x); af.u[13] = f2bf(b1.y); af.u[14] = f2bf(b1.z); af.u[15] = f2bf(b1.w);
        } else {                         // K >= 128: A = xj - xi
            const int c0 = k0 - CC;
            const float4 sa0 = *(const float4*)(xs + c0);
            const float4 sa1 = *(const float4*)(xs + c0 + 4);
            const float4 sb0 = *(const float4*)(xs + c0 + 8);
            const float4 sb1 = *(const float4*)(xs + c0 + 12);
            const float4 da0 = *(const float4*)(xd + c0);
            const float4 da1 = *(const float4*)(xd + c0 + 4);
            const float4 db0 = *(const float4*)(xd + c0 + 8);
            const float4 db1 = *(const float4*)(xd + c0 + 12);
            af.u[0] = f2bf(sa0.x - da0.x); af.u[1] = f2bf(sa0.y - da0.y);
            af.u[2] = f2bf(sa0.z - da0.z); af.u[3] = f2bf(sa0.w - da0.w);
            af.u[4] = f2bf(sa1.x - da1.x); af.u[5] = f2bf(sa1.y - da1.y);
            af.u[6] = f2bf(sa1.z - da1.z); af.u[7] = f2bf(sa1.w - da1.w);
            af.u[8]  = f2bf(sb0.x - db0.x); af.u[9]  = f2bf(sb0.y - db0.y);
            af.u[10] = f2bf(sb0.z - db0.z); af.u[11] = f2bf(sb0.w - db0.w);
            af.u[12] = f2bf(sb1.x - db1.x); af.u[13] = f2bf(sb1.y - db1.y);
            af.u[14] = f2bf(sb1.z - db1.z); af.u[15] = f2bf(sb1.w - db1.w);
        }
        // preload all 8 B fragments (16x ds_load_b128), then 8 WMMAs back-to-back
        BF16Frag bfr[8];
#pragma unroll
        for (int nt = 0; nt < 8; ++nt) {
            const uint4* bp = (const uint4*)(w1l + ((((kt << 3) + nt) << 9) + (lane << 4)));
            bfr[nt].q[0] = bp[0];
            bfr[nt].q[1] = bp[1];
        }
#pragma unroll
        for (int nt = 0; nt < 8; ++nt)
            acc[nt] = wmma_bf16(af.v, bfr[nt].v, acc[nt]);
    }

    // scale+shift+relu -> h1 (bf16) staged in this wave's LDS slice
    unsigned short* hl = h1l + (wave << 11);
#pragma unroll
    for (int nt = 0; nt < 8; ++nt) {
        const int n = (nt << 4) + col;
        const float sc = s1[n], sh = t1[n];
#pragma unroll
        for (int r = 0; r < 8; ++r) {
            float v = fmaxf(acc[nt][r] * sc + sh, 0.0f);
            hl[((r0 + r) << 7) + n] = f2bf(v);
        }
    }
    __syncthreads();

    // ---- fc2: [16 x 128] @ [128 x 128] ----
    v8f acc2[8];
#pragma unroll
    for (int i = 0; i < 8; ++i)
#pragma unroll
        for (int q = 0; q < 8; ++q) acc2[i][q] = 0.0f;

#pragma unroll
    for (int kt = 0; kt < 4; ++kt) {
        const int k0 = (kt << 5) + kb;
        BF16Frag af;
        af.q[0] = *(const uint4*)(hl + (m << 7) + k0);
        af.q[1] = *(const uint4*)(hl + (m << 7) + k0 + 8);
        BF16Frag bfr[8];
#pragma unroll
        for (int nt = 0; nt < 8; ++nt) {
            const uint4* bp = (const uint4*)(w2l + ((((kt << 3) + nt) << 9) + (lane << 4)));
            bfr[nt].q[0] = bp[0];
            bfr[nt].q[1] = bp[1];
        }
#pragma unroll
        for (int nt = 0; nt < 8; ++nt)
            acc2[nt] = wmma_bf16(af.v, bfr[nt].v, acc2[nt]);
    }

    // scale+shift+relu and scatter-sum into agg[dst]
#pragma unroll
    for (int nt = 0; nt < 8; ++nt) {
        const int n = (nt << 4) + col;
        const float sc = s2[n], sh = t2[n];
#pragma unroll
        for (int r = 0; r < 8; ++r) {
            const int ee = eBase + r0 + r;
            if (ee < E) {
                float v = fmaxf(acc2[nt][r] * sc + sh, 0.0f);
                atomicAdd(&agg[(size_t)dst[ee] * CC + n], v);
            }
        }
    }
}

// ---------------------------------------------------------------------------
// Elementwise / aggregation kernels
// ---------------------------------------------------------------------------

__global__ void ggc_agg_kernel(const float* __restrict__ mm, const int* __restrict__ src,
                               const int* __restrict__ dst, const float* __restrict__ ewb,
                               float* __restrict__ agg, int E) {
    int idx = blockIdx.x * blockDim.x + threadIdx.x;
    if (idx >= E * 32) return;
    int e  = idx >> 5;
    int cg = (idx & 31) << 2;
    float w = ewb[0];
    const float4 mv = *(const float4*)(mm + (size_t)src[e] * CC + cg);
    float* ap = agg + (size_t)dst[e] * CC + cg;
    atomicAdd(ap + 0, w * mv.x);
    atomicAdd(ap + 1, w * mv.y);
    atomicAdd(ap + 2, w * mv.z);
    atomicAdd(ap + 3, w * mv.w);
}

__global__ void gru_kernel(const float* __restrict__ gi, const float* __restrict__ gh,
                           float* __restrict__ h, int total) {
    int idx = blockIdx.x * blockDim.x + threadIdx.x;
    if (idx >= total) return;
    int n = idx >> 7, c = idx & 127;
    const float* gin = gi + (size_t)n * 384;
    const float* ghn = gh + (size_t)n * 384;
    float r  = 1.0f / (1.0f + expf(-(gin[c]       + ghn[c])));
    float z  = 1.0f / (1.0f + expf(-(gin[128 + c] + ghn[128 + c])));
    float nv = tanhf(gin[256 + c] + r * ghn[256 + c]);
    h[idx] = (1.0f - z) * nv + z * h[idx];
}

__global__ void bn_relu_kernel(const float* __restrict__ in, const float* __restrict__ s,
                               const float* __restrict__ t, float* __restrict__ out, int total) {
    int idx = blockIdx.x * blockDim.x + threadIdx.x;
    if (idx >= total) return;
    int c = idx & 127;
    out[idx] = fmaxf(in[idx] * s[c] + t[c], 0.0f);
}

__global__ void resid_bn_relu_kernel(const float* __restrict__ agg, const float* __restrict__ invd,
                                     const float* __restrict__ hprev, const float* __restrict__ s,
                                     const float* __restrict__ t, float* __restrict__ out, int total) {
    int idx = blockIdx.x * blockDim.x + threadIdx.x;
    if (idx >= total) return;
    int c = idx & 127, n = idx >> 7;
    float v = agg[idx] * invd[n] + hprev[idx];
    out[idx] = fmaxf(v * s[c] + t[c], 0.0f);
}

__global__ void final_fc_kernel(const float* __restrict__ agg, const float* __restrict__ invd,
                                const float* __restrict__ h3, const float* __restrict__ fcW,
                                const float* __restrict__ fcb, float* __restrict__ out, int Nn) {
    int n = blockIdx.x * blockDim.x + threadIdx.x;
    if (n >= Nn) return;
    float a0 = fcb[0], a1 = fcb[1];
    const float iv = invd[n];
    const float* ap = agg + (size_t)n * CC;
    const float* hp = h3  + (size_t)n * CC;
    for (int k = 0; k < CC; ++k) {
        float v = ap[k] * iv + hp[k];
        a0 += v * fcW[2 * k];
        a1 += v * fcW[2 * k + 1];
    }
    out[2 * n]     = a0;
    out[2 * n + 1] = a1;
}

// ---------------------------------------------------------------------------
// Launcher
// ---------------------------------------------------------------------------
extern "C" void kernel_launch(void* const* d_in, const int* in_sizes, int n_in,
                              void* d_out, int out_size, void* d_ws, size_t ws_size,
                              hipStream_t stream) {
    (void)in_sizes; (void)n_in; (void)out_size; (void)ws_size;

    auto F = [&](int i) { return (const float*)d_in[i]; };
    const float* x    = F(0);
    const int*   src  = (const int*)d_in[1];
    const int*   dstp = src + NE;
    const float* avW  = F(2);
    const float* avb  = F(3);
    const float* ewb  = F(5);
    const float* ggcW = F(6);
    const float* Wih  = F(7);
    const float* Whh  = F(8);
    const float* bih  = F(9);
    const float* bhh  = F(10);
    const float* fcW  = F(11);
    const float* fcb  = F(12);
    // bn1:13-16, bn2:17-20, bn3:21-24; ec bases: 25, 37, 49

    // ---- workspace carve ----
    char* w = (char*)d_ws;
    size_t off = 0;
    auto alloc = [&](size_t bytes) -> void* {
        void* p = w + off;
        off += (bytes + 255) & ~(size_t)255;
        return p;
    };
    float* deg  = (float*)alloc((size_t)NN * 4);
    float* invd = (float*)alloc((size_t)NN * 4);
    float* hA   = (float*)alloc((size_t)NN * CC * 4);
    float* hB   = (float*)alloc((size_t)NN * CC * 4);
    float* agg  = (float*)alloc((size_t)NN * CC * 4);
    float* mbuf = (float*)alloc((size_t)NN * CC * 4);
    float* gi   = (float*)alloc((size_t)NN * 384 * 4);
    float* gh   = (float*)alloc((size_t)NN * 384 * 4);
    unsigned short* avWp   = (unsigned short*)alloc(256 * 128 * 2);
    unsigned short* ggcp0  = (unsigned short*)alloc(128 * 128 * 2);
    unsigned short* ggcp1  = (unsigned short*)alloc(128 * 128 * 2);
    unsigned short* WihTp  = (unsigned short*)alloc(128 * 384 * 2);
    unsigned short* WhhTp  = (unsigned short*)alloc(128 * 384 * 2);
    unsigned short* ecW1p[3], *ecW2p[3];
    float *ecAs[3], *ecAt[3], *ecBs[3], *ecBt[3];
    for (int i = 0; i < 3; ++i) {
        ecW1p[i] = (unsigned short*)alloc(256 * 128 * 2);
        ecW2p[i] = (unsigned short*)alloc(128 * 128 * 2);
        ecAs[i] = (float*)alloc(128 * 4); ecAt[i] = (float*)alloc(128 * 4);
        ecBs[i] = (float*)alloc(128 * 4); ecBt[i] = (float*)alloc(128 * 4);
    }
    float* bnS[3]; float* bnT[3];   // bn1, bn2, bn3
    for (int i = 0; i < 3; ++i) { bnS[i] = (float*)alloc(128 * 4); bnT[i] = (float*)alloc(128 * 4); }

    // ---- prep ----
    hipMemsetAsync(deg, 0, (size_t)NN * 4, stream);
    deg_kernel<<<(NE + 255) / 256, 256, 0, stream>>>(dstp, deg, NE);
    invdeg_kernel<<<(NN + 255) / 256, 256, 0, stream>>>(deg, invd, NN);

    auto packB = [&](const float* s, unsigned short* d, int K, int N, int tr) {
        pack_b_kernel<<<(K * N + 255) / 256, 256, 0, stream>>>(s, d, K, N, tr);
    };
    packB(avW, avWp, 256, 128, 0);
    packB(ggcW,            ggcp0, 128, 128, 0);
    packB(ggcW + 128*128,  ggcp1, 128, 128, 0);
    packB(Wih, WihTp, 128, 384, 1);
    packB(Whh, WhhTp, 128, 384, 1);
    const int ecBase[3] = {25, 37, 49};
    for (int i = 0; i < 3; ++i) {
        int b = ecBase[i];
        packB(F(b + 0), ecW1p[i], 256, 128, 0);   // fc1_W
        packB(F(b + 6), ecW2p[i], 128, 128, 0);   // fc2_W
        bnfuse_kernel<<<1, 128, 0, stream>>>(F(b+2), F(b+3), F(b+4), F(b+5), F(b+1), ecAs[i], ecAt[i], 128);
        bnfuse_kernel<<<1, 128, 0, stream>>>(F(b+8), F(b+9), F(b+10), F(b+11), F(b+7), ecBs[i], ecBt[i], 128);
    }
    const int bnBase[3] = {13, 17, 21};
    for (int i = 0; i < 3; ++i)
        bnfuse_kernel<<<1, 128, 0, stream>>>(F(bnBase[i]), F(bnBase[i]+1), F(bnBase[i]+2), F(bnBase[i]+3),
                                             nullptr, bnS[i], bnT[i], 128);

    const int gemmBlocks = (NN + 15) / 16;

    // ---- av_fusion: h = x[N,256] @ av_W + av_b ----
    wmma_gemm_kernel<1><<<gemmBlocks, 256, 16 * 256 * 2, stream>>>(
        x, nullptr, avWp, nullptr, avb, hA, NN, 256, 128);

    float* h  = hA;
    float* hn = hB;

    // ---- GatedGraphConv, 2 layers ----
    unsigned short* ggcp[2] = {ggcp0, ggcp1};
    for (int l = 0; l < 2; ++l) {
        wmma_gemm_kernel<1><<<gemmBlocks, 256, 16 * 128 * 2, stream>>>(
            h, nullptr, ggcp[l], nullptr, nullptr, mbuf, NN, 128, 128);
        hipMemsetAsync(agg, 0, (size_t)NN * CC * 4, stream);
        ggc_agg_kernel<<<(NE * 32 + 255) / 256, 256, 0, stream>>>(mbuf, src, dstp, ewb, agg, NE);
        // gi = (agg * inv_deg) @ Wih^T + bih ; gh = h @ Whh^T + bhh
        wmma_gemm_kernel<3><<<gemmBlocks, 256, 16 * 128 * 2, stream>>>(
            agg, invd, WihTp, nullptr, bih, gi, NN, 128, 384);
        wmma_gemm_kernel<3><<<gemmBlocks, 256, 16 * 128 * 2, stream>>>(
            h, nullptr, WhhTp, nullptr, bhh, gh, NN, 128, 384);
        gru_kernel<<<(NN * CC + 255) / 256, 256, 0, stream>>>(gi, gh, h, NN * CC);
    }

    // ---- bn1 + relu ----
    bn_relu_kernel<<<(NN * CC + 255) / 256, 256, 0, stream>>>(h, bnS[0], bnT[0], h, NN * CC);

    // ---- EdgeConv x3 ----
    for (int li = 0; li < 3; ++li) {
        hipMemsetAsync(agg, 0, (size_t)NN * CC * 4, stream);
        edgeconv_kernel<<<(NE + 127) / 128, 256, 131072, stream>>>(
            h, src, dstp, ecW1p[li], ecAs[li], ecAt[li], ecW2p[li], ecBs[li], ecBt[li], agg, NE);
        if (li < 2) {
            resid_bn_relu_kernel<<<(NN * CC + 255) / 256, 256, 0, stream>>>(
                agg, invd, h, bnS[li + 1], bnT[li + 1], hn, NN * CC);
            float* tsw = h; h = hn; hn = tsw;
        } else {
            final_fc_kernel<<<(NN + 255) / 256, 256, 0, stream>>>(
                agg, invd, h, fcW, fcb, (float*)d_out, NN);
        }
    }
}
